// WindowedCAETransformerBlock_26706106647172
// MI455X (gfx1250) — compile-verified, matrix operand order checked by reading, not run on hardware
//
#include <hip/hip_runtime.h>

// ---------------- problem constants ----------------
#define NTOK 3136          // 16*196 merged tokens
#define MPAD 3200          // 25 * 128 padded rows
#define CDIM 768
#define N3   2304          // 3*C
#define HID  3072
#define NHEAD 12
#define HD    64
#define QSCALE 0.125f      // 64^-0.5

typedef __bf16 bf16;
typedef __attribute__((ext_vector_type(16))) __bf16 v16bf;
typedef __attribute__((ext_vector_type(8)))  float  v8f;
typedef __attribute__((ext_vector_type(4)))  unsigned int u32x4;
typedef __attribute__((ext_vector_type(8)))  int i32x8;
typedef __attribute__((ext_vector_type(4)))  int i32x4;

enum { EPI_QKV = 0, EPI_PROJ = 1, EPI_GELU = 2, EPI_OUT = 3 };

__device__ __forceinline__ v8f zero8() {
  v8f z = {0.f,0.f,0.f,0.f,0.f,0.f,0.f,0.f};
  return z;
}

__device__ __forceinline__ v8f wmma_bf16(v16bf a, v16bf b, v8f c) {
  // D = A(16x32) * B(32x16) + C(16x16 f32)
  return __builtin_amdgcn_wmma_f32_16x16x32_bf16(false, a, false, b, (short)0, c, false, false);
}

// Build a 16x32 bf16 A/B fragment for one lane from two contiguous 8-element
// (16B) chunks.  ISA layout: lane<16 holds K=[0,8)+[16,24), lane>=16 holds
// K=[8,16)+[24,32); base0 = (lane>=16)*8.
__device__ __forceinline__ v16bf load_frag(const bf16* p0, const bf16* p1) {
  union U { int4 v; bf16 h[8]; } a, b;
  a.v = *reinterpret_cast<const int4*>(p0);
  b.v = *reinterpret_cast<const int4*>(p1);
  v16bf r;
#pragma unroll
  for (int i = 0; i < 8; ++i) { r[i] = a.h[i]; r[i + 8] = b.h[i]; }
  return r;
}

// ---------------- small helper kernels ----------------
__global__ void build_qkv_bias_kernel(const float* __restrict__ qb,
                                      const float* __restrict__ vb,
                                      float* __restrict__ out) {
  int i = blockIdx.x * 256 + threadIdx.x;
  if (i < N3) out[i] = (i < CDIM) ? qb[i] : (i < 2 * CDIM ? 0.f : vb[i - 2 * CDIM]);
}

__global__ void cvt_f32_bf16_kernel(const float* __restrict__ in,
                                    bf16* __restrict__ out, int n) {
  for (int i = blockIdx.x * 256 + threadIdx.x; i < n; i += gridDim.x * 256)
    out[i] = (bf16)in[i];
}

__global__ void zero_bf16_kernel(bf16* __restrict__ p, int n) {
  for (int i = blockIdx.x * 256 + threadIdx.x; i < n; i += gridDim.x * 256)
    p[i] = (bf16)0.f;
}

// LayerNorm over C=768, one row per block, fp32 in -> bf16 out.
__global__ __launch_bounds__(256) void layernorm_bf16_kernel(
    const float* __restrict__ x, const float* __restrict__ s,
    const float* __restrict__ b, bf16* __restrict__ out) {
  __shared__ float red[256];
  const float* row = x + (size_t)blockIdx.x * CDIM;
  float sum = 0.f, sq = 0.f;
  for (int i = threadIdx.x; i < CDIM; i += 256) {
    float v = row[i];
    sum += v; sq += v * v;
  }
  red[threadIdx.x] = sum; __syncthreads();
  for (int o = 128; o > 0; o >>= 1) {
    if (threadIdx.x < o) red[threadIdx.x] += red[threadIdx.x + o];
    __syncthreads();
  }
  float mu = red[0] / CDIM;
  __syncthreads();
  red[threadIdx.x] = sq; __syncthreads();
  for (int o = 128; o > 0; o >>= 1) {
    if (threadIdx.x < o) red[threadIdx.x] += red[threadIdx.x + o];
    __syncthreads();
  }
  float var = red[0] / CDIM - mu * mu;
  float inv = rsqrtf(var + 1e-5f);
  bf16* orow = out + (size_t)blockIdx.x * CDIM;
  for (int i = threadIdx.x; i < CDIM; i += 256)
    orow[i] = (bf16)((row[i] - mu) * inv * s[i] + b[i]);
}

// ---------------- generic bf16 WMMA GEMM with fused epilogues ----------------
// OUT[M,N] = A[Mpad,K] @ W[N,K]^T (+ epilogue).
// Block tile 128x256, K-step 32.  8 waves as 2(m) x 4(n); each wave owns a
// 64x64 tile = 16 WMMA accumulators.  B tile (256x32) is staged into LDS by
// the Tensor Data Mover (one descriptor per K-step, issued by wave 0); the A
// tile goes through the regular vector-load path.
#define BM 128
#define BN 256
#define BK 32

template <int EPI>
__global__ __launch_bounds__(256) void gemm_bf16_kernel(
    const bf16* __restrict__ A, const bf16* __restrict__ W,
    int M, int N, int K,
    const float* __restrict__ bias, const float* __restrict__ gamma,
    const float* __restrict__ resid, float* __restrict__ outF,
    bf16* __restrict__ outH,
    bf16* __restrict__ qb, bf16* __restrict__ kb, bf16* __restrict__ vtb) {
  __shared__ __align__(16) bf16 As[BM][BK];
  __shared__ __align__(16) bf16 Bs[BN][BK];

  const int tid  = threadIdx.x;
  const int lane = tid & 31;
  const int wave = tid >> 5;
  const int wm   = wave >> 2;      // 0..1
  const int wn   = wave & 3;       // 0..3
  const int hi   = lane >> 4;      // 0/1
  const int l15  = lane & 15;
  const int base0 = hi * 8;

  const int m0 = blockIdx.x * BM;
  const int n0 = blockIdx.y * BN;

  v8f acc[4][4];
#pragma unroll
  for (int i = 0; i < 4; ++i)
#pragma unroll
    for (int j = 0; j < 4; ++j) acc[i][j] = zero8();

  const unsigned ldsB = (unsigned)(size_t)(&Bs[0][0]);  // LDS byte offset

  for (int k0 = 0; k0 < K; k0 += BK) {
    // ---- B tile 256x32 via Tensor Data Mover (wave 0 issues, EXEC ignored)
    if (tid < 32) {
      unsigned long long ga =
          (unsigned long long)(size_t)(W + (size_t)n0 * K + k0);
      u32x4 g0;
      g0[0] = 1u;                                   // count=1, user mode
      g0[1] = ldsB;                                 // lds_addr
      g0[2] = (unsigned)ga;                         // global_addr[31:0]
      g0[3] = (unsigned)((ga >> 32) & 0x01FFFFFFu) | (2u << 30);  // type=2
      i32x8 g1;
      g1[0] = 1 << 16;                              // data_size=1 (2 bytes)
      g1[1] = (int)(((unsigned)K & 0xFFFFu) << 16); // tensor_dim0 lo16
      g1[2] = (int)((((unsigned)K >> 16) & 0xFFFFu) |
                    (((unsigned)N & 0xFFFFu) << 16));  // dim0 hi16 | dim1 lo16
      g1[3] = (int)((((unsigned)N >> 16) & 0xFFFFu) |
                    ((unsigned)BK << 16));          // dim1 hi16 | tile_dim0=32
      g1[4] = BN;                                   // tile_dim1=256
      g1[5] = K;                                    // tensor_dim0_stride lo32
      g1[6] = 0;
      g1[7] = 0;
      i32x4 g2 = {0, 0, 0, 0};
      i32x4 g3 = {0, 0, 0, 0};
      i32x8 g4 = {0, 0, 0, 0, 0, 0, 0, 0};
      __builtin_amdgcn_tensor_load_to_lds(g0, g1, g2, g3, g4, 0);
    }

    // ---- A tile 128x32 via vector loads: 512 x 16B chunks, 2 per thread
#pragma unroll
    for (int i = 0; i < 2; ++i) {
      int chunk = tid * 2 + i;
      int r = chunk >> 2, cc = (chunk & 3) * 8;
      *reinterpret_cast<int4*>(&As[r][cc]) =
          *reinterpret_cast<const int4*>(&A[(size_t)(m0 + r) * K + k0 + cc]);
    }
    if (k0 + BK < K)
      __builtin_prefetch(&A[(size_t)(m0 + (tid >> 1)) * K + k0 + BK], 0, 1);

    if (tid < 32) __builtin_amdgcn_s_wait_tensorcnt(0);
    __syncthreads();

    v16bf af[4], bfr[4];
#pragma unroll
    for (int mt = 0; mt < 4; ++mt) {
      int r = wm * 64 + mt * 16 + l15;
      af[mt] = load_frag(&As[r][base0], &As[r][base0 + 16]);
    }
#pragma unroll
    for (int nt = 0; nt < 4; ++nt) {
      int r = wn * 64 + nt * 16 + l15;
      bfr[nt] = load_frag(&Bs[r][base0], &Bs[r][base0 + 16]);
    }
#pragma unroll
    for (int mt = 0; mt < 4; ++mt)
#pragma unroll
      for (int nt = 0; nt < 4; ++nt)
        acc[mt][nt] = wmma_bf16(af[mt], bfr[nt], acc[mt][nt]);
    __syncthreads();
  }

  // epilogue: C layout -> lane = n (l15), vgpr r -> m = r + 8*hi
#pragma unroll
  for (int mt = 0; mt < 4; ++mt) {
#pragma unroll
    for (int nt = 0; nt < 4; ++nt) {
#pragma unroll
      for (int r = 0; r < 8; ++r) {
        int m = m0 + wm * 64 + mt * 16 + r + hi * 8;
        int n = n0 + wn * 64 + nt * 16 + l15;
        float v = acc[mt][nt][r];
        if (EPI == EPI_QKV) {
          if (m < M) {
            v += bias[n];
            int which = n / CDIM, c = n % CDIM;
            int h = c >> 6, d = c & 63;
            if (which == 0)
              qb[((size_t)h * NTOK + m) * HD + d] = (bf16)(v * QSCALE);
            else if (which == 1)
              kb[((size_t)h * NTOK + m) * HD + d] = (bf16)v;
            else
              vtb[((size_t)h * HD + d) * NTOK + m] = (bf16)v;
          }
        } else if (EPI == EPI_PROJ) {
          if (m < M)
            outF[(size_t)m * CDIM + n] =
                resid[(size_t)m * CDIM + n] + gamma[n] * (v + bias[n]);
        } else if (EPI == EPI_GELU) {
          float t = v + bias[n];
          float g = 0.5f * t * (1.f + erff(t * 0.70710678118654752f));
          outH[(size_t)m * HID + n] = (bf16)g;   // pad rows finite, discarded
        } else {  // EPI_OUT
          if (m < M)
            outF[(size_t)m * CDIM + n] =
                resid[(size_t)m * CDIM + n] + gamma[n] * (v + bias[n]);
        }
      }
    }
  }
}

// ---------------- flash attention (one pass, online softmax) ----------------
// grid(49, 12), 128 threads = 4 waves; each wave owns 16 query rows and scans
// keys in steps of 64 (4 score tiles -> 8 QK^T WMMA + 8 PV WMMA per step).
__global__ __launch_bounds__(128) void attn_kernel(
    const bf16* __restrict__ qB, const bf16* __restrict__ kB,
    const bf16* __restrict__ vT, const unsigned char* __restrict__ mask,
    bf16* __restrict__ out /* [MPAD][CDIM] bf16 */) {
  __shared__ __align__(16) bf16 Pl[4][16][64];

  const int lane = threadIdx.x & 31;
  const int wave = threadIdx.x >> 5;
  const int h = blockIdx.y;
  const int qbase = (blockIdx.x * 4 + wave) * 16;   // 196 tiles exactly
  const int hi = lane >> 4, l15 = lane & 15, base0 = hi * 8;

  const bf16* qh = qB + (size_t)h * NTOK * HD;
  const bf16* kh = kB + (size_t)h * NTOK * HD;
  const bf16* vh = vT + (size_t)h * HD * NTOK;

  // Q fragments: 16 rows x 64 (two K=32 steps); q pre-scaled by 1/8
  const size_t qoff = (size_t)(qbase + l15) * HD;
  v16bf qf[2];
  qf[0] = load_frag(qh + qoff + base0,      qh + qoff + base0 + 16);
  qf[1] = load_frag(qh + qoff + 32 + base0, qh + qoff + 32 + base0 + 16);

  float mrow[8], lrow[8];
  v8f accO[4];
#pragma unroll
  for (int r = 0; r < 8; ++r) { mrow[r] = -3.0e38f; lrow[r] = 0.f; }
#pragma unroll
  for (int t = 0; t < 4; ++t) accO[t] = zero8();

  for (int kb0 = 0; kb0 < NTOK; kb0 += 64) {
    // ---- S = q @ k^T for 64 keys (four 16-wide N tiles) ----
    v8f s[4];
#pragma unroll
    for (int jt = 0; jt < 4; ++jt) s[jt] = zero8();
#pragma unroll
    for (int jt = 0; jt < 4; ++jt) {
      size_t koff = (size_t)(kb0 + jt * 16 + l15) * HD;
      v16bf b0 = load_frag(kh + koff + base0,      kh + koff + base0 + 16);
      v16bf b1 = load_frag(kh + koff + 32 + base0, kh + koff + 32 + base0 + 16);
      s[jt] = wmma_bf16(qf[0], b0, s[jt]);
      s[jt] = wmma_bf16(qf[1], b1, s[jt]);
    }

    // ---- mask + online softmax (per-row stats; lane = key col) ----
    float mk[4];
#pragma unroll
    for (int jt = 0; jt < 4; ++jt)
      mk[jt] = mask[kb0 + jt * 16 + l15] ? -1.0e30f : 0.f;

    float p[4][8];
#pragma unroll
    for (int r = 0; r < 8; ++r) {
#pragma unroll
      for (int jt = 0; jt < 4; ++jt) p[jt][r] = s[jt][r] + mk[jt];
      float mx = fmaxf(fmaxf(p[0][r], p[1][r]), fmaxf(p[2][r], p[3][r]));
#pragma unroll
      for (int o = 8; o >= 1; o >>= 1) mx = fmaxf(mx, __shfl_xor(mx, o, 32));
      float mnew = fmaxf(mrow[r], mx);
      float alpha = __expf(mrow[r] - mnew);
      mrow[r] = mnew;
#pragma unroll
      for (int jt = 0; jt < 4; ++jt) p[jt][r] = __expf(p[jt][r] - mnew);
      float ps = (p[0][r] + p[1][r]) + (p[2][r] + p[3][r]);
#pragma unroll
      for (int o = 8; o >= 1; o >>= 1) ps += __shfl_xor(ps, o, 32);
      lrow[r] = lrow[r] * alpha + ps;
#pragma unroll
      for (int t = 0; t < 4; ++t) accO[t][r] *= alpha;
    }

    // ---- relayout P: C layout -> row-major LDS tile -> A fragments ----
#pragma unroll
    for (int r = 0; r < 8; ++r) {
      int m = r + hi * 8;
#pragma unroll
      for (int jt = 0; jt < 4; ++jt)
        Pl[wave][m][jt * 16 + l15] = (bf16)p[jt][r];
    }
    __syncthreads();
    v16bf pf0 = load_frag(&Pl[wave][l15][base0],      &Pl[wave][l15][base0 + 16]);
    v16bf pf1 = load_frag(&Pl[wave][l15][32 + base0], &Pl[wave][l15][32 + base0 + 16]);
    __syncthreads();

    // ---- O += P @ V  (V stored transposed: B operand row-major [d][key]) ----
#pragma unroll
    for (int dt = 0; dt < 4; ++dt) {
      size_t voff = (size_t)(dt * 16 + l15) * NTOK + kb0;
      v16bf bv0 = load_frag(vh + voff + base0,      vh + voff + base0 + 16);
      v16bf bv1 = load_frag(vh + voff + 32 + base0, vh + voff + 32 + base0 + 16);
      accO[dt] = wmma_bf16(pf0, bv0, accO[dt]);
      accO[dt] = wmma_bf16(pf1, bv1, accO[dt]);
    }
  }

  // ---- normalize + store bf16 into [token][C] at this head's columns ----
#pragma unroll
  for (int dt = 0; dt < 4; ++dt) {
#pragma unroll
    for (int r = 0; r < 8; ++r) {
      int m = qbase + r + hi * 8;
      int c = h * HD + dt * 16 + l15;
      out[(size_t)m * CDIM + c] = (bf16)(accO[dt][r] / lrow[r]);
    }
  }
}

// ---------------- host orchestration ----------------
extern "C" void kernel_launch(void* const* d_in, const int* in_sizes, int n_in,
                              void* d_out, int out_size, void* d_ws, size_t ws_size,
                              hipStream_t stream) {
  (void)in_sizes; (void)n_in; (void)out_size; (void)ws_size;
  const float* x       = (const float*)d_in[0];
  const unsigned char* mask = (const unsigned char*)d_in[1];
  const float* qkv_w   = (const float*)d_in[2];
  const float* q_bias  = (const float*)d_in[3];
  const float* v_bias  = (const float*)d_in[4];
  const float* proj_w  = (const float*)d_in[5];
  const float* proj_b  = (const float*)d_in[6];
  const float* n1s     = (const float*)d_in[7];
  const float* n1b     = (const float*)d_in[8];
  const float* n2s     = (const float*)d_in[9];
  const float* n2b     = (const float*)d_in[10];
  const float* mlp_w1  = (const float*)d_in[11];
  const float* mlp_b1  = (const float*)d_in[12];
  const float* mlp_w2  = (const float*)d_in[13];
  const float* mlp_b2  = (const float*)d_in[14];
  const float* g1      = (const float*)d_in[15];
  const float* g2      = (const float*)d_in[16];
  float* outp = (float*)d_out;

  char* ws = (char*)d_ws;
  size_t off = 0;
  auto alloc = [&](size_t bytes) -> void* {
    void* p = ws + off;
    off = (off + bytes + 255) & ~(size_t)255;
    return p;
  };
  float* qkvBias = (float*)alloc((size_t)N3 * 4);
  bf16* xn    = (bf16*)alloc((size_t)MPAD * CDIM * 2);
  bf16* wqkv  = (bf16*)alloc((size_t)N3 * CDIM * 2);
  bf16* wproj = (bf16*)alloc((size_t)CDIM * CDIM * 2);
  bf16* w1h   = (bf16*)alloc((size_t)HID * CDIM * 2);
  bf16* w2h   = (bf16*)alloc((size_t)CDIM * HID * 2);
  bf16* qbuf  = (bf16*)alloc((size_t)NHEAD * NTOK * HD * 2);
  bf16* kbuf  = (bf16*)alloc((size_t)NHEAD * NTOK * HD * 2);
  bf16* vtbuf = (bf16*)alloc((size_t)NHEAD * HD * NTOK * 2);
  bf16* attn  = (bf16*)alloc((size_t)MPAD * CDIM * 2);
  float* x2   = (float*)alloc((size_t)MPAD * CDIM * 4);
  bf16* hn    = (bf16*)alloc((size_t)MPAD * CDIM * 2);
  bf16* h1    = (bf16*)alloc((size_t)MPAD * HID * 2);

  auto cvt = [&](const float* src, bf16* dst, int n) {
    int g = (n + 255) / 256; if (g > 2048) g = 2048;
    cvt_f32_bf16_kernel<<<g, 256, 0, stream>>>(src, dst, n);
  };

  // weight / bias prep (deterministic every call)
  build_qkv_bias_kernel<<<(N3 + 255) / 256, 256, 0, stream>>>(q_bias, v_bias, qkvBias);
  cvt(qkv_w,  wqkv,  N3 * CDIM);
  cvt(proj_w, wproj, CDIM * CDIM);
  cvt(mlp_w1, w1h,   HID * CDIM);
  cvt(mlp_w2, w2h,   CDIM * HID);

  // zero the 64 pad rows of every GEMM A-side activation buffer
  const int padN = (MPAD - NTOK) * CDIM;  // 49152
  zero_bf16_kernel<<<(padN + 255) / 256, 256, 0, stream>>>(xn + (size_t)NTOK * CDIM, padN);
  zero_bf16_kernel<<<(padN + 255) / 256, 256, 0, stream>>>(hn + (size_t)NTOK * CDIM, padN);
  zero_bf16_kernel<<<(padN + 255) / 256, 256, 0, stream>>>(attn + (size_t)NTOK * CDIM, padN);

  // 1) LN1 -> bf16
  layernorm_bf16_kernel<<<NTOK, 256, 0, stream>>>(x, n1s, n1b, xn);

  // 2) QKV gemm, fused bias + q-scale + scatter into q / k / v^T
  gemm_bf16_kernel<EPI_QKV><<<dim3(MPAD / BM, N3 / BN), 256, 0, stream>>>(
      xn, wqkv, NTOK, N3, CDIM, qkvBias, nullptr, nullptr, nullptr, nullptr,
      qbuf, kbuf, vtbuf);

  // 3) flash attention
  attn_kernel<<<dim3(49, NHEAD), 128, 0, stream>>>(qbuf, kbuf, vtbuf, mask, attn);

  // 4) projection + gamma_1 residual -> x2 (fp32)
  gemm_bf16_kernel<EPI_PROJ><<<dim3(MPAD / BM, CDIM / BN), 256, 0, stream>>>(
      attn, wproj, NTOK, CDIM, CDIM, proj_b, g1, x, x2, nullptr,
      nullptr, nullptr, nullptr);

  // 5) LN2 -> bf16
  layernorm_bf16_kernel<<<NTOK, 256, 0, stream>>>(x2, n2s, n2b, hn);

  // 6) MLP up + exact GELU -> bf16
  gemm_bf16_kernel<EPI_GELU><<<dim3(MPAD / BM, HID / BN), 256, 0, stream>>>(
      hn, w1h, NTOK, HID, CDIM, mlp_b1, nullptr, nullptr, nullptr, h1,
      nullptr, nullptr, nullptr);

  // 7) MLP down + gamma_2 residual -> d_out (fp32)
  gemm_bf16_kernel<EPI_OUT><<<dim3(MPAD / BM, CDIM / BN), 256, 0, stream>>>(
      h1, w2h, NTOK, CDIM, HID, mlp_b2, g2, x2, outp, nullptr,
      nullptr, nullptr, nullptr);
}
